// ChunkedMultiHeadCardPassingLayer_2774548873905
// MI455X (gfx1250) — matrix operand order repaired and checked by reading, not compiled, not run on hardware
//
#include <hip/hip_runtime.h>
#include <hip/hip_bf16.h>
#include <cmath>

// ---------------------------------------------------------------------------
// ChunkedMultiHeadCardPassingLayer for MI455X (gfx1250, wave32, WMMA).
// All big GEMMs run as bf16 x bf16 -> f32 via v_wmma_f32_16x16x32_bf16,
// with GEMM tiles staged LDS-side through the CDNA5 async copy path
// (global_load_async_to_lds_b128 + s_wait_asynccnt), double-buffered so the
// DMA of tile k+1 overlaps the WMMAs of tile k.
// Problem sizes fixed by the reference: B=4 T=4096 C=2048 H=16 D=128.
// ---------------------------------------------------------------------------

#define NUMH   16
#define SCHUNK 128
#define BB     4
#define TTOK   4096
#define CCH    2048
#define DDIM   128
#define NCH    32                       // TTOK / SCHUNK
#define MROWS  (BB*TTOK)                // 16384
#define MMLP   (BB*NUMH*NCH*SCHUNK)     // 262144
#define EPSF   1e-5f

typedef __attribute__((ext_vector_type(16))) __bf16 bf16x16;
typedef __attribute__((ext_vector_type(8)))  __bf16 bf16x8;
typedef __attribute__((ext_vector_type(8)))  float  f32x8;

__device__ __forceinline__ unsigned short f2bf(float f) {
  unsigned u = __float_as_uint(f);
  u += 0x7fffu + ((u >> 16) & 1u);      // round-to-nearest-even
  return (unsigned short)(u >> 16);
}

// ---------------------------------------------------------------------------
// Generic 128x128-tile WMMA GEMM:  C = A(MxK, bf16 row-major) *
//   Bt(NxK, bf16 N-major i.e. pre-transposed weights) + bias.
// 8 waves (2x4), each wave computes a 64x32 region = 4x2 WMMA 16x16 tiles.
// MODE 0: f32 out.  MODE 1: exact GELU -> bf16.  MODE 2: bias -> bf16 with
// (b,h,n,s)-row -> (b,t,c)-row permutation (MLP second GEMM epilogue).
// ---------------------------------------------------------------------------
template<int MODE>
__global__ __launch_bounds__(256)
void wmma_gemm(const unsigned short* __restrict__ A,
               const unsigned short* __restrict__ Bt,
               const float* __restrict__ bias,
               void* __restrict__ Cout,
               int M, int N, int K)
{
  // Double-buffered staging, padded row stride (40 shorts = 80B) for banks.
  __shared__ unsigned short AsB[2][128 * 40];
  __shared__ unsigned short BsB[2][128 * 40];

  const int tid   = threadIdx.x;
  const int lane  = tid & 31;
  const int wave  = tid >> 5;
  const int waveM = wave >> 2;              // 0..1
  const int waveN = wave & 3;               // 0..3
  const int l15   = lane & 15;
  const int lh    = lane >> 4;              // lane half

  const int m0 = blockIdx.y * 128;
  const int n0 = blockIdx.x * 128;

  const int lrow  = tid >> 1;               // 0..127 : tile row/col to stage
  const int lhalf = tid & 1;                // which 16-element K chunk

  const unsigned short* gA = A  + (size_t)(m0 + lrow) * K + lhalf * 16;
  const unsigned short* gB = Bt + (size_t)(n0 + lrow) * K + lhalf * 16;

  // LDS byte offsets for this thread's 32B staging chunk (generic-pointer low
  // 32 bits == LDS offset per the gfx1250 aperture rules).
  unsigned ldsA[2], ldsB[2];
  ldsA[0] = (unsigned)(size_t)&AsB[0][lrow * 40 + lhalf * 16];
  ldsA[1] = (unsigned)(size_t)&AsB[1][lrow * 40 + lhalf * 16];
  ldsB[0] = (unsigned)(size_t)&BsB[0][lrow * 40 + lhalf * 16];
  ldsB[1] = (unsigned)(size_t)&BsB[1][lrow * 40 + lhalf * 16];

  // Async DMA of one 128x32 bf16 tile pair into LDS buffer `buf`.
  // offset:16 applies to BOTH the LDS and global address (ISA 10.7 pseudocode)
  auto issue_async = [&](int buf, int k0) {
    unsigned long long ga = (unsigned long long)(size_t)(gA + k0);
    unsigned long long gb = (unsigned long long)(size_t)(gB + k0);
    asm volatile(
        "global_load_async_to_lds_b128 %0, %2, off\n\t"
        "global_load_async_to_lds_b128 %0, %2, off offset:16\n\t"
        "global_load_async_to_lds_b128 %1, %3, off\n\t"
        "global_load_async_to_lds_b128 %1, %3, off offset:16"
        :: "v"(ldsA[buf]), "v"(ldsB[buf]), "v"(ga), "v"(gb)
        : "memory");
  };

  f32x8 acc[4][2];
  const f32x8 z = {0.f, 0.f, 0.f, 0.f, 0.f, 0.f, 0.f, 0.f};
#pragma unroll
  for (int mt = 0; mt < 4; ++mt)
#pragma unroll
    for (int nt = 0; nt < 2; ++nt) acc[mt][nt] = z;

  // Prologue: fill buffer 0.
  issue_async(0, 0);
  asm volatile("s_wait_asynccnt 0x0" ::: "memory");
  __syncthreads();

  int cur = 0;
  for (int k0 = 0; k0 < K; k0 += 32) {
    // Kick the DMA for the next tile while we do math on this one.
    if (k0 + 32 < K) {
      issue_async(cur ^ 1, k0 + 32);
      if (k0 + 64 < K) {                    // warm L2 one tile further out
        __builtin_prefetch(gA + k0 + 64, 0, 1);
        __builtin_prefetch(gB + k0 + 64, 0, 1);
      }
    }

    const unsigned short* Ab = AsB[cur];
    const unsigned short* Bb = BsB[cur];

    // Fragment layouts per CDNA5 ISA 7.12.2 (16-bit A 16x32 / B 32x16).
    union { bf16x16 v; bf16x8 h[2]; } af[4], bfr[2];
#pragma unroll
    for (int mt = 0; mt < 4; ++mt) {
      const unsigned short* ap = &Ab[(waveM * 64 + mt * 16 + l15) * 40 + 8 * lh];
      af[mt].h[0] = *(const bf16x8*)(ap);        // K = 8*lh .. 8*lh+7
      af[mt].h[1] = *(const bf16x8*)(ap + 16);   // K = 16+8*lh ..
    }
#pragma unroll
    for (int nt = 0; nt < 2; ++nt) {
      const unsigned short* bp = &Bb[(waveN * 32 + nt * 16 + l15) * 40 + 8 * lh];
      bfr[nt].h[0] = *(const bf16x8*)(bp);       // K = 8*lh .. 8*lh+7
      bfr[nt].h[1] = *(const bf16x8*)(bp + 16);  // K = 16+8*lh ..
    }
#pragma unroll
    for (int mt = 0; mt < 4; ++mt)
#pragma unroll
      for (int nt = 0; nt < 2; ++nt)
        acc[mt][nt] = __builtin_amdgcn_wmma_f32_16x16x32_bf16(
            false, af[mt].v, false, bfr[nt].v, (short)0, acc[mt][nt],
            false, false);

    // Next buffer's DMA done + every wave finished reading this buffer
    // (each wave's ds_loads retired before its WMMAs -> before this barrier).
    asm volatile("s_wait_asynccnt 0x0" ::: "memory");
    __syncthreads();
    cur ^= 1;
  }

  // Epilogue. C/D layout: VGPR r, lane l -> row = r + 8*(l>>4), col = l&15.
#pragma unroll
  for (int mt = 0; mt < 4; ++mt) {
#pragma unroll
    for (int nt = 0; nt < 2; ++nt) {
      const int gc  = n0 + waveN * 32 + nt * 16 + l15;
      const float bv = bias[gc];
      const int grb = m0 + waveM * 64 + mt * 16 + 8 * lh;
      f32x8 a = acc[mt][nt];
#pragma unroll
      for (int r = 0; r < 8; ++r) {
        const int gr = grb + r;
        float v = a[r] + bv;
        if (MODE == 0) {
          ((float*)Cout)[(size_t)gr * N + gc] = v;
        } else if (MODE == 1) {
          v = 0.5f * v * (1.0f + erff(v * 0.70710678118654752f));
          ((unsigned short*)Cout)[(size_t)gr * N + gc] = f2bf(v);
        } else {
          // row gr = ((b*16+h)*32+n)*128+s  ->  out[b][n*128+s][h*128+gc]
          const int s  = gr & (SCHUNK - 1);
          const int nn = (gr >> 7) & (NCH - 1);
          const int hh = (gr >> 12) & (NUMH - 1);
          const int bb = gr >> 16;
          const size_t dst =
              (((size_t)bb * TTOK + (size_t)nn * SCHUNK + s) * CCH) +
              (size_t)hh * DDIM + gc;
          ((unsigned short*)Cout)[dst] = f2bf(v);
        }
      }
    }
  }
}

// ---------------------------------------------------------------------------
// f32 -> bf16 elementwise (x4 vectorized)
// ---------------------------------------------------------------------------
__global__ __launch_bounds__(256)
void cvt_bf16x4(const float* __restrict__ in, unsigned short* __restrict__ out,
                long n4)
{
  long i = (long)blockIdx.x * 256 + threadIdx.x;
  if (i >= n4) return;
  float4 v = ((const float4*)in)[i];
  uint2 p;
  p.x = (unsigned)f2bf(v.x) | ((unsigned)f2bf(v.y) << 16);
  p.y = (unsigned)f2bf(v.z) | ((unsigned)f2bf(v.w) << 16);
  ((uint2*)out)[i] = p;
}

// Wt[n*K + k] = bf16(W[k*N + n])   (tiled transpose through LDS)
__global__ __launch_bounds__(256)
void transpose_cvt(const float* __restrict__ W, unsigned short* __restrict__ Wt,
                   int K, int N)
{
  __shared__ float tile[32][33];
  const int kb = blockIdx.y * 32, nb = blockIdx.x * 32;
  const int tx = threadIdx.x & 31, ty = threadIdx.x >> 5;  // 8 rows per pass
#pragma unroll
  for (int i = 0; i < 32; i += 8)
    tile[ty + i][tx] = W[(size_t)(kb + ty + i) * N + nb + tx];
  __syncthreads();
#pragma unroll
  for (int i = 0; i < 32; i += 8)
    Wt[(size_t)(nb + ty + i) * K + kb + tx] = f2bf(tile[tx][ty + i]);
}

// gated = sigmoid(gates) * marks   (written back into marks buffer)
__global__ __launch_bounds__(256)
void gated_k(float* __restrict__ marks, const float* __restrict__ gates, long n4)
{
  long i = (long)blockIdx.x * 256 + threadIdx.x;
  if (i >= n4) return;
  float4 m = ((const float4*)marks)[i];
  float4 g = ((const float4*)gates)[i];
  m.x *= 1.f / (1.f + expf(-g.x));
  m.y *= 1.f / (1.f + expf(-g.y));
  m.z *= 1.f / (1.f + expf(-g.z));
  m.w *= 1.f / (1.f + expf(-g.w));
  ((float4*)marks)[i] = m;
}

// chunk_sums[b,h,n,d] = sum_s gated[b, n*S+s, h*D+d]
__global__ __launch_bounds__(256)
void chunk_sums_k(const float* __restrict__ gated, float* __restrict__ sums)
{
  int idx = blockIdx.x * 256 + threadIdx.x;     // 262144
  int d = idx & 127, n = (idx >> 7) & 31, h = (idx >> 12) & 15, b = idx >> 16;
  const float* p = gated + ((size_t)b * TTOK + (size_t)n * SCHUNK) * CCH +
                   (size_t)h * DDIM + d;
  float s = 0.f;
  for (int ss = 0; ss < SCHUNK; ++ss) s += p[(size_t)ss * CCH];
  sums[idx] = s;
}

// carries[b,h,n,d] = exclusive prefix over n of chunk_sums
__global__ __launch_bounds__(128)
void carries_k(const float* __restrict__ sums, float* __restrict__ carr)
{
  const int bh = blockIdx.x;                    // 64
  const int d  = threadIdx.x;
  const float* cs = sums + (size_t)bh * NCH * DDIM + d;
  float* co = carr + (size_t)bh * NCH * DDIM + d;
  float run = 0.f;
  for (int n = 0; n < NCH; ++n) { co[(size_t)n * DDIM] = run; run += cs[(size_t)n * DDIM]; }
}

// sum + sumsq reduction across a 128-thread (4-wave) block
__device__ __forceinline__ void block_red2(float& s, float& q, float* red)
{
#pragma unroll
  for (int o = 16; o > 0; o >>= 1) {
    s += __shfl_xor(s, o, 32);
    q += __shfl_xor(q, o, 32);
  }
  const int w = threadIdx.x >> 5;
  if ((threadIdx.x & 31) == 0) { red[w] = s; red[4 + w] = q; }
  __syncthreads();
  s = red[0] + red[1] + red[2] + red[3];
  q = red[4] + red[5] + red[6] + red[7];
  __syncthreads();
}

// One block per (b,h,n): nc = LN(carry); then for each s:
//   card = LN(nc + local_cumsum_exclusive(s)); emit combined = [x | card] bf16
__global__ __launch_bounds__(128)
void pass2_k(const float* __restrict__ gated, const float* __restrict__ carr,
             const unsigned short* __restrict__ xb,
             const float* __restrict__ carry_g, const float* __restrict__ carry_b,
             const float* __restrict__ card_g,  const float* __restrict__ card_b,
             unsigned short* __restrict__ combined)
{
  __shared__ float red[8];
  const int bhn = blockIdx.x;                   // 2048 = ((b*16+h)*32+n)
  const int d = threadIdx.x;
  const int n = bhn & 31, h = (bhn >> 5) & 15, b = bhn >> 9;

  float cv = carr[(size_t)bhn * DDIM + d];
  float s = cv, q = cv * cv;
  block_red2(s, q, red);
  float mu = s * (1.f / DDIM), var = q * (1.f / DDIM) - mu * mu;
  const float nc = (cv - mu) * rsqrtf(var + EPSF) * carry_g[d] + carry_b[d];

  const float cgd = card_g[d], cbd = card_b[d];
  const float* gp = gated + ((size_t)b * TTOK + (size_t)n * SCHUNK) * CCH +
                    (size_t)h * DDIM + d;
  const unsigned short* xp = xb + ((size_t)b * TTOK + (size_t)n * SCHUNK) * CCH +
                             (size_t)h * DDIM + d;
  unsigned short* cp = combined + (size_t)bhn * SCHUNK * 256 + d;

  float run = 0.f;                              // exclusive local cumsum
  for (int ss = 0; ss < SCHUNK; ++ss) {
    float cl = nc + run;
    float s2 = cl, q2 = cl * cl;
    block_red2(s2, q2, red);
    float mu2 = s2 * (1.f / DDIM), var2 = q2 * (1.f / DDIM) - mu2 * mu2;
    float card = (cl - mu2) * rsqrtf(var2 + EPSF) * cgd + cbd;
    cp[(size_t)ss * 256]       = xp[(size_t)ss * CCH];
    cp[(size_t)ss * 256 + 128] = f2bf(card);
    run += gp[(size_t)ss * CCH];
  }
}

// out = x + LN(out) over C=2048, in place, one block per token row
__global__ __launch_bounds__(256)
void final_ln_k(const float* __restrict__ x, const float* __restrict__ lng,
                const float* __restrict__ lnb, float* __restrict__ out)
{
  __shared__ float red[16];
  const int row = blockIdx.x;
  float* op = out + (size_t)row * CCH;
  float vals[8], s = 0.f, q = 0.f;
#pragma unroll
  for (int i = 0; i < 8; ++i) {
    float v = op[threadIdx.x + i * 256];
    vals[i] = v; s += v; q += v * v;
  }
#pragma unroll
  for (int o = 16; o > 0; o >>= 1) { s += __shfl_xor(s, o, 32); q += __shfl_xor(q, o, 32); }
  const int w = threadIdx.x >> 5;
  if ((threadIdx.x & 31) == 0) { red[w] = s; red[8 + w] = q; }
  __syncthreads();
  if (threadIdx.x == 0) {
    float ts = 0.f, tq = 0.f;
    for (int i = 0; i < 8; ++i) { ts += red[i]; tq += red[8 + i]; }
    red[0] = ts; red[8] = tq;
  }
  __syncthreads();
  s = red[0]; q = red[8];
  float mu = s * (1.f / CCH), var = q * (1.f / CCH) - mu * mu;
  float rs = rsqrtf(var + EPSF);
  const float* xp = x + (size_t)row * CCH;
#pragma unroll
  for (int i = 0; i < 8; ++i) {
    int c = threadIdx.x + i * 256;
    op[c] = xp[c] + (vals[i] - mu) * rs * lng[c] + lnb[c];
  }
}

// ---------------------------------------------------------------------------
extern "C" void kernel_launch(void* const* d_in, const int* in_sizes, int n_in,
                              void* d_out, int out_size, void* d_ws, size_t ws_size,
                              hipStream_t stream)
{
  (void)in_sizes; (void)n_in; (void)out_size; (void)ws_size;
  const float* x       = (const float*)d_in[0];
  const float* mark_W  = (const float*)d_in[1];
  const float* mark_b  = (const float*)d_in[2];
  const float* gate_W  = (const float*)d_in[3];
  const float* gate_b  = (const float*)d_in[4];
  const float* carry_g = (const float*)d_in[5];
  const float* carry_b = (const float*)d_in[6];
  const float* card_g  = (const float*)d_in[7];
  const float* card_b  = (const float*)d_in[8];
  const float* mlp_W1  = (const float*)d_in[9];
  const float* mlp_b1  = (const float*)d_in[10];
  const float* mlp_W2  = (const float*)d_in[11];
  const float* mlp_b2  = (const float*)d_in[12];
  const float* proj_W  = (const float*)d_in[13];
  const float* proj_b  = (const float*)d_in[14];
  const float* ln_g    = (const float*)d_in[15];
  const float* ln_b    = (const float*)d_in[16];

  char* ws = (char*)d_ws;
  size_t off = 0;
  auto bump = [&](size_t bytes) -> char* {
    char* p = ws + off; off += (bytes + 255) & ~(size_t)255; return p;
  };
  unsigned short* xb     = (unsigned short*)bump((size_t)MROWS * CCH * 2); // -> hbuf later
  unsigned short* markWt = (unsigned short*)bump((size_t)CCH * CCH * 2);
  unsigned short* gateWt = (unsigned short*)bump((size_t)CCH * CCH * 2);
  unsigned short* projWt = (unsigned short*)bump((size_t)CCH * CCH * 2);
  unsigned short* W1t    = (unsigned short*)bump((size_t)256 * 256 * 2);
  unsigned short* W2t    = (unsigned short*)bump((size_t)128 * 256 * 2);
  float* marks           = (float*)bump((size_t)MROWS * CCH * 4);          // -> gated
  float* gates           = (float*)bump((size_t)MROWS * CCH * 4);          // -> h1 (bf16)
  float* csums           = (float*)bump((size_t)BB * NUMH * NCH * DDIM * 4);
  float* carr            = (float*)bump((size_t)BB * NUMH * NCH * DDIM * 4);
  unsigned short* comb   = (unsigned short*)bump((size_t)MMLP * 256 * 2);
  unsigned short* h1     = (unsigned short*)gates;   // gates dead after gated_k
  unsigned short* hbuf   = xb;                       // xb dead after pass2_k

  // 1) precision conversions / weight transposes
  cvt_bf16x4<<<dim3(((size_t)MROWS * CCH / 4 + 255) / 256), 256, 0, stream>>>(
      x, xb, (long)MROWS * CCH / 4);
  transpose_cvt<<<dim3(CCH / 32, CCH / 32), 256, 0, stream>>>(mark_W, markWt, CCH, CCH);
  transpose_cvt<<<dim3(CCH / 32, CCH / 32), 256, 0, stream>>>(gate_W, gateWt, CCH, CCH);
  transpose_cvt<<<dim3(CCH / 32, CCH / 32), 256, 0, stream>>>(proj_W, projWt, CCH, CCH);
  transpose_cvt<<<dim3(256 / 32, 256 / 32), 256, 0, stream>>>(mlp_W1, W1t, 256, 256);
  transpose_cvt<<<dim3(128 / 32, 256 / 32), 256, 0, stream>>>(mlp_W2, W2t, 256, 128);

  // 2) marks / gates GEMMs (bf16 WMMA, f32 out)
  wmma_gemm<0><<<dim3(CCH / 128, MROWS / 128), 256, 0, stream>>>(
      xb, markWt, mark_b, marks, MROWS, CCH, CCH);
  wmma_gemm<0><<<dim3(CCH / 128, MROWS / 128), 256, 0, stream>>>(
      xb, gateWt, gate_b, gates, MROWS, CCH, CCH);

  // 3) gated = sigmoid(gates) * marks  (into marks)
  gated_k<<<dim3(((size_t)MROWS * CCH / 4 + 255) / 256), 256, 0, stream>>>(
      marks, gates, (long)MROWS * CCH / 4);

  // 4) chunk scans + carry LN + card LN -> combined bf16 [x | cards]
  chunk_sums_k<<<dim3(MMLP / SCHUNK / 2), 256, 0, stream>>>(marks, csums); // 1024 blocks
  carries_k<<<dim3(BB * NUMH), 128, 0, stream>>>(csums, carr);
  pass2_k<<<dim3(BB * NUMH * NCH), 128, 0, stream>>>(
      marks, carr, xb, carry_g, carry_b, card_g, card_b, comb);

  // 5) MLP: (M,256)x(256,256) GELU -> bf16 ; (M,256)x(256,128) -> permuted bf16
  wmma_gemm<1><<<dim3(256 / 128, MMLP / 128), 256, 0, stream>>>(
      comb, W1t, mlp_b1, h1, MMLP, 256, 256);
  wmma_gemm<2><<<dim3(128 / 128, MMLP / 128), 256, 0, stream>>>(
      h1, W2t, mlp_b2, hbuf, MMLP, 128, 256);

  // 6) projection GEMM -> f32 into d_out
  wmma_gemm<0><<<dim3(CCH / 128, MROWS / 128), 256, 0, stream>>>(
      hbuf, projWt, proj_b, d_out, MROWS, CCH, CCH);

  // 7) out = x + LN(out) in place
  final_ln_k<<<dim3(MROWS), 256, 0, stream>>>(x, ln_g, ln_b, (float*)d_out);
}
